// circ_layer_17575006175271
// MI455X (gfx1250) — matrix operand order repaired and checked by reading, not compiled, not run on hardware
//
#include <hip/hip_runtime.h>

typedef __bf16 v16bf __attribute__((ext_vector_type(16)));
typedef float  v8f   __attribute__((ext_vector_type(8)));

#define N_FEAT 1024
#define BM 128
#define BN 128
#define BK 32

__device__ __forceinline__ unsigned short f2bf(float f) {
  unsigned int u = __float_as_uint(f);
  u += 0x7FFFu + ((u >> 16) & 1u);       // round-to-nearest-even
  return (unsigned short)(u >> 16);
}
__device__ __forceinline__ float bf2f(unsigned short h) {
  return __uint_as_float(((unsigned int)h) << 16);
}
__device__ __forceinline__ unsigned int pk2(unsigned short a, unsigned short b) {
  return (unsigned int)a | ((unsigned int)b << 16);
}

// out[b,i] = mm*x + mm + x,  mm[b,i] = sum_j x[b,j] * w[(j-i) mod 1024]
__global__ __launch_bounds__(256) void circ_wmma_kernel(
    const float* __restrict__ x, const float* __restrict__ w,
    float* __restrict__ out) {
  // Double-buffered LDS tiles in WMMA *fragment order* (each lane's fragment
  // is one contiguous 32B read). A: hi/lo split of x tile; B: circulant tile.
  __shared__ unsigned short Ah[2][BM * BK], Al[2][BM * BK];
  __shared__ unsigned short Bh[2][BN * BK], Bl[2][BN * BK];
  __shared__ unsigned short w2h[2 * N_FEAT], w2l[2 * N_FEAT];

  const int tid  = threadIdx.x;
  const int lane = tid & 31;
  const int wv   = tid >> 5;                 // 8 waves; wave owns rows wv*16..+15
  const int m0   = blockIdx.x * BM;
  const int n0   = blockIdx.y * BN;

  // --- one-time: duplicated weight vector, split into bf16 hi/lo ---
  for (int t = tid; t < 2 * N_FEAT; t += 256) {
    float f = w[t & (N_FEAT - 1)];
    unsigned short h = f2bf(f);
    unsigned short l = f2bf(f - bf2f(h));
    w2h[t] = h; w2l[t] = l;
  }

  v8f acc[8] = {};                           // 8 N-subtiles of 16x16 fp32

  // staging geometry: 256 threads; thread -> (row r, k-half kh), 16 elems each
  const int r  = tid >> 1;                   // 0..127
  const int kh = tid & 1;                    // 0,1 -> k offset 0 / 16
  // A fragment-order bases (contiguous k order, k = 16*(lane/16) + idx, same as B):
  //   lane = (r&15) + 16*(k>=16),  idx = k%16
  const int baseA0 = (r >> 4) * 512 + ((r & 15) + 16 * kh) * 16;     // k%16 in 0..7
  const int baseA1 = (r >> 4) * 512 + ((r & 15) + 16 * kh) * 16 + 8; // k%16 in 8..15
  const int bbase  = r * 32 + kh * 16;       // B tile: Bt[n_local][k_local]
  const float* xrow = x + (size_t)(m0 + r) * N_FEAT + kh * 16;

  // ---- software pipeline: preload chunk 0 into registers ----
  float4 f0 = ((const float4*)xrow)[0];
  float4 f1 = ((const float4*)xrow)[1];
  float4 f2 = ((const float4*)xrow)[2];
  float4 f3 = ((const float4*)xrow)[3];

  int p = 0;
  for (int k0 = 0; k0 < N_FEAT; k0 += BK, p ^= 1) {
    // ---- convert in-register chunk to bf16 hi/lo ----
    float fv[16] = {f0.x, f0.y, f0.z, f0.w, f1.x, f1.y, f1.z, f1.w,
                    f2.x, f2.y, f2.z, f2.w, f3.x, f3.y, f3.z, f3.w};
    unsigned short hh[16], ll[16];
#pragma unroll
    for (int j = 0; j < 16; ++j) {
      hh[j] = f2bf(fv[j]);
      ll[j] = f2bf(fv[j] - bf2f(hh[j]));
    }

    // ---- issue global loads for next chunk NOW (in flight across compute) ----
    if (k0 + BK < N_FEAT) {
      const float4* nsrc = (const float4*)(xrow + k0 + BK);
      f0 = nsrc[0]; f1 = nsrc[1]; f2 = nsrc[2]; f3 = nsrc[3];
      if (k0 + 2 * BK < N_FEAT)              // keep L2 one more chunk ahead
        __builtin_prefetch((const char*)nsrc + BK * 4, 0, 0);
    }

    // ---- store A (fragment order, packed 16B writes) ----
    uint4 v;
    v.x = pk2(hh[0], hh[1]); v.y = pk2(hh[2], hh[3]);
    v.z = pk2(hh[4], hh[5]); v.w = pk2(hh[6], hh[7]);
    *(uint4*)&Ah[p][baseA0] = v;
    v.x = pk2(hh[8], hh[9]);  v.y = pk2(hh[10], hh[11]);
    v.z = pk2(hh[12], hh[13]); v.w = pk2(hh[14], hh[15]);
    *(uint4*)&Ah[p][baseA1] = v;
    v.x = pk2(ll[0], ll[1]); v.y = pk2(ll[2], ll[3]);
    v.z = pk2(ll[4], ll[5]); v.w = pk2(ll[6], ll[7]);
    *(uint4*)&Al[p][baseA0] = v;
    v.x = pk2(ll[8], ll[9]);  v.y = pk2(ll[10], ll[11]);
    v.z = pk2(ll[12], ll[13]); v.w = pk2(ll[14], ll[15]);
    *(uint4*)&Al[p][baseA1] = v;

    // ---- stage B: circulant tile Bt[n][k] = w[(k0+k - n0-n) mod 1024] ----
    {
      const int wbase = N_FEAT + k0 + kh * 16 - n0 - r;  // in [1, 2047-15]
      unsigned short bh[16], bl[16];
#pragma unroll
      for (int j = 0; j < 16; ++j) { bh[j] = w2h[wbase + j]; bl[j] = w2l[wbase + j]; }
      v.x = pk2(bh[0], bh[1]); v.y = pk2(bh[2], bh[3]);
      v.z = pk2(bh[4], bh[5]); v.w = pk2(bh[6], bh[7]);
      *(uint4*)&Bh[p][bbase] = v;
      v.x = pk2(bh[8], bh[9]);  v.y = pk2(bh[10], bh[11]);
      v.z = pk2(bh[12], bh[13]); v.w = pk2(bh[14], bh[15]);
      *(uint4*)&Bh[p][bbase + 8] = v;
      v.x = pk2(bl[0], bl[1]); v.y = pk2(bl[2], bl[3]);
      v.z = pk2(bl[4], bl[5]); v.w = pk2(bl[6], bl[7]);
      *(uint4*)&Bl[p][bbase] = v;
      v.x = pk2(bl[8], bl[9]);  v.y = pk2(bl[10], bl[11]);
      v.z = pk2(bl[12], bl[13]); v.w = pk2(bl[14], bl[15]);
      *(uint4*)&Bl[p][bbase + 8] = v;
    }

    // single barrier per chunk: safe with alternating buffers (a writer of
    // buffer p at iter i+2 has passed barrier i+1, which implies all readers
    // of buffer p from iter i are done)
    __syncthreads();

    // ---- compute: bf16x3 split-precision WMMA from buffer p ----
    const v16bf a_h = *(const v16bf*)&Ah[p][wv * 512 + lane * 16];
    const v16bf a_l = *(const v16bf*)&Al[p][wv * 512 + lane * 16];
#pragma unroll
    for (int nt = 0; nt < 8; ++nt) {
      const int boff = (nt * 16 + (lane & 15)) * 32 + (lane >> 4) * 16;
      const v16bf b_h = *(const v16bf*)&Bh[p][boff];
      const v16bf b_l = *(const v16bf*)&Bl[p][boff];
      acc[nt] = __builtin_amdgcn_wmma_f32_16x16x32_bf16(
          false, a_h, false, b_h, (short)0, acc[nt], false, false);
      acc[nt] = __builtin_amdgcn_wmma_f32_16x16x32_bf16(
          false, a_h, false, b_l, (short)0, acc[nt], false, false);
      acc[nt] = __builtin_amdgcn_wmma_f32_16x16x32_bf16(
          false, a_l, false, b_h, (short)0, acc[nt], false, false);
    }
  }

  // ---- fused epilogue: out = mm*(x+1) + x ----
  // C/D layout: lane (0-15): N=lane, VGPR v -> M=v; lanes 16-31 -> M=8+v.
  const int colb = lane & 15;
  const int rowb = m0 + wv * 16 + ((lane >> 4) << 3);
#pragma unroll
  for (int nt = 0; nt < 8; ++nt) {
    const int gc = n0 + nt * 16 + colb;
#pragma unroll
    for (int vv = 0; vv < 8; ++vv) {
      const size_t off = (size_t)(rowb + vv) * N_FEAT + gc;
      const float xv = x[off];
      out[off] = fmaf(acc[nt][vv], xv + 1.0f, xv);
    }
  }
}

extern "C" void kernel_launch(void* const* d_in, const int* in_sizes, int n_in,
                              void* d_out, int out_size, void* d_ws, size_t ws_size,
                              hipStream_t stream) {
  const float* x = (const float*)d_in[0];
  const float* w = (const float*)d_in[1];
  float* out = (float*)d_out;
  const int ntok = in_sizes[0] / N_FEAT;     // 65536
  dim3 grid(ntok / BM, N_FEAT / BN);         // (512, 8)
  circ_wmma_kernel<<<grid, dim3(256), 0, stream>>>(x, w, out);
}
